// GraphEncoder_37769942401265
// MI455X (gfx1250) — matrix operand order repaired and checked by reading, not compiled, not run on hardware
//
#include <hip/hip_runtime.h>

typedef __attribute__((ext_vector_type(8)))  __bf16 v8bf;   // 16 B -> 4 VGPRs
typedef __attribute__((ext_vector_type(16))) __bf16 v16bf;  // 32 B -> 8 VGPRs
typedef __attribute__((ext_vector_type(8)))  float  v8f;

__device__ inline v16bf cat8(v8bf a, v8bf b) {
  return __builtin_shufflevector(a, b, 0, 1, 2, 3, 4, 5, 6, 7, 8, 9, 10, 11, 12, 13, 14, 15);
}

// ---------------------------------------------------------------- utilities
__global__ void zero_f32(float* __restrict__ p, long n) {
  long i = (long)blockIdx.x * blockDim.x + threadIdx.x;
  long stride = (long)gridDim.x * blockDim.x;
  for (; i < n; i += stride) p[i] = 0.0f;
}

__global__ void degree_count(const int* __restrict__ ei, float* __restrict__ cnt, int E) {
  int e = blockIdx.x * blockDim.x + threadIdx.x;
  if (e < E) atomicAdd(&cnt[ei[E + e]], 1.0f);
}

__global__ void finalize_invdeg(float* __restrict__ d, int n) {
  int i = blockIdx.x * blockDim.x + threadIdx.x;
  if (i < n) d[i] = 1.0f / fmaxf(d[i], 1.0f);
}

// --------------------------------------------- fp32 -> bf16 hi/lo split passes
// a ~= hi + lo with lo = bf16(a - float(hi));  4 elements per thread.
__global__ void split_bf16(const float* __restrict__ in, __bf16* __restrict__ hi,
                           __bf16* __restrict__ lo, long n4) {
  long i = (long)blockIdx.x * blockDim.x + threadIdx.x;
  if (i >= n4) return;
  const float4 a = *(const float4*)(in + i * 4);
#pragma unroll
  for (int j = 0; j < 4; ++j) {
    float aj = (&a.x)[j];
    __bf16 h = (__bf16)aj;
    hi[i * 4 + j] = h;
    lo[i * 4 + j] = (__bf16)(aj - (float)h);
  }
}

// Same, with per-row scale (mean normalization of agg): row = idx >> shift.
__global__ void split_bf16_scaled(const float* __restrict__ in, const float* __restrict__ rowscale,
                                  __bf16* __restrict__ hi, __bf16* __restrict__ lo,
                                  long n4, int shift) {
  long i = (long)blockIdx.x * blockDim.x + threadIdx.x;
  if (i >= n4) return;
  float s = rowscale[(i * 4) >> shift];
  const float4 a = *(const float4*)(in + i * 4);
#pragma unroll
  for (int j = 0; j < 4; ++j) {
    float aj = (&a.x)[j] * s;
    __bf16 h = (__bf16)aj;
    hi[i * 4 + j] = h;
    lo[i * 4 + j] = (__bf16)(aj - (float)h);
  }
}

// ------------------------------------------------- edge scatter (mean numerator)
// agg (<=41 MB) fits the 192 MB L2 -> global_atomic_add_f32 resolves in L2.
__global__ void scatter_mean_accum(const float* __restrict__ x, const int* __restrict__ ei,
                                   float* __restrict__ agg, int E, int din) {
  int chunks = din >> 2;
  long tid = (long)blockIdx.x * blockDim.x + threadIdx.x;
  if (tid >= (long)E * chunks) return;
  int e = (int)(tid / chunks);
  int c = (int)(tid - (long)e * chunks);
  int s = ei[e];
  int d = ei[E + e];
  const float4 v = *(const float4*)(x + (size_t)s * din + (size_t)c * 4);
  float* dp = agg + (size_t)d * din + (size_t)c * 4;
  atomicAdd(dp + 0, v.x);
  atomicAdd(dp + 1, v.y);
  atomicAdd(dp + 2, v.z);
  atomicAdd(dp + 3, v.w);
}

// ------------------------------------------------- bf16x3 WMMA accumulation
// Operands pre-split to bf16 hi/lo; loads land directly in WMMA layout:
//  A (16x32, 16-bit, ISA 7.12.2): VGPR v<4 = K(hi*8+2v, +1); v>=4 = K(16+hi*8+...).
//    -> two contiguous 16B v8bf loads per part.
//  B (32x16): lane col n=lane&15, same K order as A: v<4 = K(hi*8+2v, +1); v>=4 = K(16+hi*8+...)
//    -> two contiguous 16B v8bf loads per part from weight row W[n][*].
__device__ inline v8f mm_accum(v8f acc,
                               const __bf16* __restrict__ aH, const __bf16* __restrict__ aL,
                               const __bf16* __restrict__ wH, const __bf16* __restrict__ wL,
                               int din, int hi) {
#pragma unroll
  for (int kk = 0; kk < din; kk += 32) {
    v16bf ah = cat8(*(const v8bf*)(aH + kk + hi * 8), *(const v8bf*)(aH + kk + 16 + hi * 8));
    v16bf al = cat8(*(const v8bf*)(aL + kk + hi * 8), *(const v8bf*)(aL + kk + 16 + hi * 8));
    v16bf bh = cat8(*(const v8bf*)(wH + kk + hi * 8), *(const v8bf*)(wH + kk + 16 + hi * 8));
    v16bf bl = cat8(*(const v8bf*)(wL + kk + hi * 8), *(const v8bf*)(wL + kk + 16 + hi * 8));
    acc = __builtin_amdgcn_wmma_f32_16x16x32_bf16(false, ah, false, bh, (short)0, acc, false, false);
    acc = __builtin_amdgcn_wmma_f32_16x16x32_bf16(false, ah, false, bl, (short)0, acc, false, false);
    acc = __builtin_amdgcn_wmma_f32_16x16x32_bf16(false, al, false, bh, (short)0, acc, false, false);
  }
  return acc;
}

// ------------------------------------------------- fused SAGE layer GEMM
// out = relu?( aggN @ Wl.T + x @ Wr.T + b ); one wave per 16x16 tile.
// N=40000 -> 2500 exact M-tiles, dout=256 -> 16 N-tiles; no partial tiles,
// EXEC stays all-1s in every wave that reaches the WMMAs.
template <int DIN>
__global__ void __launch_bounds__(256)
sage_gemm_fused(const __bf16* __restrict__ aggH, const __bf16* __restrict__ aggL,
                const __bf16* __restrict__ xH,   const __bf16* __restrict__ xL,
                const __bf16* __restrict__ wlH,  const __bf16* __restrict__ wlL,
                const __bf16* __restrict__ wrH,  const __bf16* __restrict__ wrL,
                const float* __restrict__ bias, float* __restrict__ out,
                int dout, int mtiles, int do_relu) {
  int wave = (blockIdx.x * blockDim.x + threadIdx.x) >> 5;
  int ntiles = dout >> 4;
  if (wave >= mtiles * ntiles) return;  // wave-uniform guard
  int lane = threadIdx.x & 31;
  int nt = wave % ntiles;
  int mt = wave / ntiles;
  int m  = lane & 15;
  int hi = lane >> 4;
  size_t arow = (size_t)(mt * 16 + m) * DIN;
  size_t wrow = (size_t)(nt * 16 + m) * DIN;

  v8f acc = {};
  acc = mm_accum(acc, aggH + arow, aggL + arow, wlH + wrow, wlL + wrow, DIN, hi);
  acc = mm_accum(acc, xH + arow,   xL + arow,   wrH + wrow, wrL + wrow, DIN, hi);

  int ncol = nt * 16 + m;
  float bn = bias[ncol];
#pragma unroll
  for (int v = 0; v < 8; ++v) {
    float val = acc[v] + bn;
    if (do_relu) val = fmaxf(val, 0.0f);
    out[(size_t)(mt * 16 + hi * 8 + v) * dout + ncol] = val;  // C/D: row = v + 8*hi
  }
}

// ---------------------------------------------------------------- launcher
extern "C" void kernel_launch(void* const* d_in, const int* in_sizes, int n_in,
                              void* d_out, int out_size, void* d_ws, size_t ws_size,
                              hipStream_t stream) {
  const float* x   = (const float*)d_in[0];
  const int*   ei  = (const int*)d_in[1];  // edge_index [2, E]
  const float* W[3][2] = {{(const float*)d_in[2], (const float*)d_in[3]},
                          {(const float*)d_in[5], (const float*)d_in[6]},
                          {(const float*)d_in[8], (const float*)d_in[9]}};
  const float* B[3] = {(const float*)d_in[4], (const float*)d_in[7], (const float*)d_in[10]};

  const int IN_DIM = 128, HID_DIM = 256;
  const int N = in_sizes[0] / IN_DIM;  // 40000
  const int E = in_sizes[1] / 2;       // 640000

  // workspace carve-out (~206 MB)
  char* w = (char*)d_ws;
  auto take = [&](size_t bytes) { char* p = w; w += (bytes + 255) & ~(size_t)255; return p; };
  float*  inv_deg = (float*)take((size_t)N * 4);
  float*  aggF    = (float*)take((size_t)N * HID_DIM * 4);
  float*  h1      = (float*)take((size_t)N * HID_DIM * 4);
  float*  h2      = (float*)take((size_t)N * HID_DIM * 4);
  __bf16* aggH    = (__bf16*)take((size_t)N * HID_DIM * 2);
  __bf16* aggL    = (__bf16*)take((size_t)N * HID_DIM * 2);
  __bf16* xH      = (__bf16*)take((size_t)N * HID_DIM * 2);
  __bf16* xL      = (__bf16*)take((size_t)N * HID_DIM * 2);
  __bf16* wlH     = (__bf16*)take((size_t)HID_DIM * HID_DIM * 2);
  __bf16* wlL     = (__bf16*)take((size_t)HID_DIM * HID_DIM * 2);
  __bf16* wrH     = (__bf16*)take((size_t)HID_DIM * HID_DIM * 2);
  __bf16* wrL     = (__bf16*)take((size_t)HID_DIM * HID_DIM * 2);

  // degrees once (same edge list for all layers)
  zero_f32<<<1024, 256, 0, stream>>>(inv_deg, (long)N);
  degree_count<<<(E + 255) / 256, 256, 0, stream>>>(ei, inv_deg, E);
  finalize_invdeg<<<(N + 255) / 256, 256, 0, stream>>>(inv_deg, N);

  int mtiles = N / 16;  // 2500
  auto run_layer = [&](const float* xin, int din, const float* Wl, const float* Wr,
                       const float* b, float* out, int relu) {
    long nelem = (long)N * din;
    int shift = (din == 128) ? 7 : 8;
    // aggregate
    zero_f32<<<2048, 256, 0, stream>>>(aggF, nelem);
    long sthreads = (long)E * (din >> 2);
    scatter_mean_accum<<<(int)((sthreads + 255) / 256), 256, 0, stream>>>(xin, ei, aggF, E, din);
    // pre-split operands into bf16 hi/lo (memory-bound, once per layer)
    split_bf16_scaled<<<(int)((nelem / 4 + 255) / 256), 256, 0, stream>>>(aggF, inv_deg, aggH, aggL, nelem / 4, shift);
    split_bf16<<<(int)((nelem / 4 + 255) / 256), 256, 0, stream>>>(xin, xH, xL, nelem / 4);
    long wn4 = (long)HID_DIM * din / 4;
    split_bf16<<<(int)((wn4 + 255) / 256), 256, 0, stream>>>(Wl, wlH, wlL, wn4);
    split_bf16<<<(int)((wn4 + 255) / 256), 256, 0, stream>>>(Wr, wrH, wrL, wn4);
    // fused GEMM
    int tiles = mtiles * (HID_DIM / 16);
    int blocks = (tiles * 32 + 255) / 256;
    if (din == 128)
      sage_gemm_fused<128><<<blocks, 256, 0, stream>>>(aggH, aggL, xH, xL, wlH, wlL, wrH, wrL,
                                                       b, out, HID_DIM, mtiles, relu);
    else
      sage_gemm_fused<256><<<blocks, 256, 0, stream>>>(aggH, aggL, xH, xL, wlH, wlL, wrH, wrL,
                                                       b, out, HID_DIM, mtiles, relu);
  };

  run_layer(x,  IN_DIM,  W[0][0], W[0][1], B[0], h1, 1);
  run_layer(h1, HID_DIM, W[1][0], W[1][1], B[1], h2, 1);
  run_layer(h2, HID_DIM, W[2][0], W[2][1], B[2], (float*)d_out, 0);
}
